// KVMem_76716705841579
// MI455X (gfx1250) — matrix operand order repaired and checked by reading, not compiled, not run on hardware
//
#include <hip/hip_runtime.h>

// ---------------- CDNA5 (gfx1250) vector types ----------------
typedef __attribute__((ext_vector_type(16))) __bf16        v16bf;
typedef __attribute__((ext_vector_type(8)))  float         v8f;
typedef __attribute__((ext_vector_type(4)))  float         f32x4;
typedef __attribute__((ext_vector_type(4)))  unsigned int  u32x4;

union V16 {
  v16bf    v;
  __bf16   e[16];
  unsigned d[8];
  u32x4    u[2];
};
union V32 {
  __bf16 e[32];
  u32x4  u[4];
};
union PK8 {
  __bf16   e[8];
  unsigned u[4];
};

// ---------------- problem constants ----------------
#define MEMDIM   1024
#define MEMSIZE  32768
#define NHEADS   8
#define HD       128      // per-head dim
#define NTOK     1024     // B*S

#define MTILE    16       // tokens per wave
#define MG       4        // M-group waves per block -> 64 tokens/block
#define ZG       2        // Z-split groups per block
#define NWAVES   (MG*ZG)  // 8 waves, 256 threads
#define ZPW      (MEMSIZE/ZG)   // 16384 slots per z-group
#define ZC       32       // z-chunk per iteration (== WMMA K)
#define NIT      (ZPW/ZC)       // 512 iterations

// LDS strides (bf16 units), padded for bank-conflict-free fragment reads
#define KSTR     136            // K tile row: [z][d], 128 + 8
#define KZG      (32*KSTR)
#define VSTR     40             // V tile row: [d][z], 32 + 8
#define VZG      (128*VSTR)

#define V_OFF      (2*KZG*2)               // 17,408 bytes into a buffer
#define BUF_BYTES  (V_OFF + 2*VZG*2)       // 37,888 per buffer
#define SMEM_BYTES (2*BUF_BYTES)           // 75,776 (double buffered)
// reduction overlay inside buffer 0 (used only after the loop's final barrier)
#define OSTR     132
#define ORED_OFF 0                         // 4 slots * 16*132*4 = 33,792 B
#define MRED_OFF (4*MTILE*OSTR*4)
#define LRED_OFF (MRED_OFF + 4*MTILE*4)    // ends 34,304 < 37,888

__device__ inline void cvt4(__bf16* d, f32x4 f) {
  d[0] = (__bf16)f.x; d[1] = (__bf16)f.y; d[2] = (__bf16)f.z; d[3] = (__bf16)f.w;
}

struct Stage { f32x4 r[16]; };   // 64 floats per thread per chunk

__global__ __launch_bounds__(256) void kvmem_attn_kernel(
    const float* __restrict__ h,        // [NTOK, MEMDIM]
    const float* __restrict__ keys_w,   // [MEMSIZE, MEMDIM]
    const float* __restrict__ values_w, // [MEMDIM, MEMSIZE]
    float*       __restrict__ out)      // [NTOK, MEMDIM]
{
  __shared__ __align__(16) unsigned char smem[SMEM_BYTES];

  const int tid   = threadIdx.x;
  const int w     = tid >> 5;
  const int lane  = tid & 31;
  const int col   = lane & 15;      // token column of this lane (N dim of S^T/O^T)
  const int hi    = lane >> 4;      // half-group
  const int kbase = hi * 16;        // B operand: K rows kbase..kbase+15
  const int ko    = hi * 8;         // unused; A operands also read K rows kbase..kbase+15

  const int mg = w & (MG - 1);
  const int zg = w >> 2;

  const int head = blockIdx.x >> 4;        // head slow -> L2 weight sharing
  const int tile = blockIdx.x & 15;
  const int t0   = tile * (MG * MTILE) + mg * MTILE;

  const float* kwp = keys_w   + head * HD;
  const float* vwp = values_w + (size_t)head * HD * MEMSIZE;

  // staging role (whole block stages both z-group chunks)
  const int s_zg  = tid >> 7;
  const int s_tt  = tid & 127;
  const int s_z   = s_tt >> 2;
  const int s_seg = s_tt & 3;
  const int s_d   = s_tt;
  const float* kStagePtr = kwp + (size_t)(s_zg * ZPW + s_z) * MEMDIM + s_seg * 32;
  const float* vStagePtr = vwp + (size_t)s_d * MEMSIZE + s_zg * ZPW;
  const int kDstOff = s_zg * KZG + s_z * KSTR + s_seg * 32;   // bf16 elems
  const int vDstOff = s_zg * VZG + s_d * VSTR;
  // fragment read offsets (bf16 elems) inside a buffer
  const int kFragOff = zg * KZG + col * KSTR;
  const int vFragOff = zg * VZG + col * VSTR;

  // ---- Q as 4 stationary B-operand fragments (16 contiguous d per lane)
  v16bf bQ[4];
  {
    const float* qbase = h + (size_t)(t0 + col) * MEMDIM + head * HD + kbase;
    #pragma unroll
    for (int i = 0; i < 4; ++i) {
      const f32x4* q4 = (const f32x4*)(qbase + i * 32);
      V16 r;
      cvt4(r.e + 0,  q4[0]); cvt4(r.e + 4,  q4[1]);
      cvt4(r.e + 8,  q4[2]); cvt4(r.e + 12, q4[3]);
      bQ[i] = r.v;
    }
  }

  // ---- O^T accumulators (8 d-tiles; lane column = token) + per-token stats
  v8f o[8];
  #pragma unroll
  for (int dt = 0; dt < 8; ++dt) { v8f z = {0.f,0.f,0.f,0.f,0.f,0.f,0.f,0.f}; o[dt] = z; }
  float mrow = -3.0e38f, lrow = 0.0f;

  // ---- prologue: stage chunk 0 into buffer 0
  Stage st;
  {
    #pragma unroll
    for (int i = 0; i < 8; ++i) st.r[i]     = ((const f32x4*)kStagePtr)[i];
    #pragma unroll
    for (int i = 0; i < 8; ++i) st.r[8 + i] = ((const f32x4*)vStagePtr)[i];
    V32 kc, vc;
    #pragma unroll
    for (int i = 0; i < 8; ++i) { cvt4(kc.e + 4*i, st.r[i]); cvt4(vc.e + 4*i, st.r[8+i]); }
    u32x4* kd = (u32x4*)((__bf16*)smem + kDstOff);
    u32x4* vd = (u32x4*)((__bf16*)(smem + V_OFF) + vDstOff);
    #pragma unroll
    for (int j = 0; j < 4; ++j) kd[j] = kc.u[j];
    #pragma unroll
    for (int j = 0; j < 4; ++j) vd[j] = vc.u[j];
  }
  __syncthreads();

  #pragma unroll 1
  for (int it = 0; it < NIT; ++it) {
    unsigned char* rb = smem + (size_t)(it & 1) * BUF_BYTES;        // read buffer
    unsigned char* wb = smem + (size_t)((it + 1) & 1) * BUF_BYTES;  // write buffer
    const __bf16* myK = (const __bf16*)rb + kFragOff;
    const __bf16* myV = (const __bf16*)(rb + V_OFF) + vFragOff;

    // ---- issue global loads for chunk it+1 (overlapped with score WMMAs)
    const bool more = (it + 1) < NIT;
    if (more) {
      const float* kp = kStagePtr + (size_t)(it + 1) * ZC;
      const float* vp = vStagePtr + (size_t)(it + 1) * ZC;
      #pragma unroll
      for (int i = 0; i < 8; ++i) st.r[i]     = ((const f32x4*)kp)[i];
      #pragma unroll
      for (int i = 0; i < 8; ++i) st.r[8 + i] = ((const f32x4*)vp)[i];
    }
    if ((it + 2) < NIT) {                   // CDNA5 global_prefetch_b8
      __builtin_prefetch(kStagePtr + (size_t)(it + 2) * ZC, 0, 1);
      __builtin_prefetch(vStagePtr + (size_t)(it + 2) * ZC, 0, 1);
    }

    // ---- S^T = K * Q^T : two 16x16 tiles (z rows 0..15 / 16..31)
    v8f c0 = {0.f,0.f,0.f,0.f,0.f,0.f,0.f,0.f};
    v8f c1 = {0.f,0.f,0.f,0.f,0.f,0.f,0.f,0.f};
    #pragma unroll
    for (int i = 0; i < 4; ++i) {
      const __bf16* k0 = myK + i * 32;
      const __bf16* k1 = myK + 16 * KSTR + i * 32;
      V16 a0, a1;
      a0.u[0] = *(const u32x4*)(k0 + kbase); a0.u[1] = *(const u32x4*)(k0 + kbase + 8);
      a1.u[0] = *(const u32x4*)(k1 + kbase); a1.u[1] = *(const u32x4*)(k1 + kbase + 8);
      c0 = __builtin_amdgcn_wmma_f32_16x16x32_bf16(false, a0.v, false, bQ[i],
                                                   (short)0, c0, false, false);
      c1 = __builtin_amdgcn_wmma_f32_16x16x32_bf16(false, a1.v, false, bQ[i],
                                                   (short)0, c1, false, false);
    }

    // ---- convert + store chunk it+1 into the other buffer (st dies here)
    if (more) {
      V32 kc, vc;
      #pragma unroll
      for (int i = 0; i < 8; ++i) { cvt4(kc.e + 4*i, st.r[i]); cvt4(vc.e + 4*i, st.r[8+i]); }
      u32x4* kd = (u32x4*)((__bf16*)wb + kDstOff);
      u32x4* vd = (u32x4*)((__bf16*)(wb + V_OFF) + vDstOff);
      #pragma unroll
      for (int j = 0; j < 4; ++j) kd[j] = kc.u[j];
      #pragma unroll
      for (int j = 0; j < 4; ++j) vd[j] = vc.u[j];
    }

    // ---- per-token online softmax: in-lane reduce + one xor-16 exchange
    float mx = fmaxf(c0[0], c1[0]);
    #pragma unroll
    for (int r = 1; r < 8; ++r) mx = fmaxf(mx, fmaxf(c0[r], c1[r]));
    mx = fmaxf(mx, __shfl_xor(mx, 16, 32));
    const float mnew = fmaxf(mrow, mx);
    float p0[8], p1[8], s = 0.0f;
    #pragma unroll
    for (int r = 0; r < 8; ++r) {
      p0[r] = __expf(c0[r] - mnew);
      p1[r] = __expf(c1[r] - mnew);
      s += p0[r] + p1[r];
    }
    s += __shfl_xor(s, 16, 32);
    // rescale only when some lane's running max actually moved (wave-uniform skip)
    if (__ballot(mx > mrow)) {
      const float scale = __expf(mrow - mnew);
      lrow *= scale;
      #pragma unroll
      for (int dt = 0; dt < 8; ++dt) o[dt] = o[dt] * scale;   // packed f32 muls
    }
    lrow += s;
    mrow = mnew;

    // ---- build P^T B-operand in registers: swap 4 packed dwords across halves
    PK8 t0p, t1p;
    #pragma unroll
    for (int r = 0; r < 8; ++r) { t0p.e[r] = (__bf16)p0[r]; t1p.e[r] = (__bf16)p1[r]; }
    V16 bp;
    #pragma unroll
    for (int j = 0; j < 4; ++j) {
      unsigned send = hi ? t0p.u[j] : t1p.u[j];
      unsigned rcv  = (unsigned)__shfl_xor((int)send, 16, 32);
      bp.d[j]     = hi ? rcv      : t0p.u[j];
      bp.d[4 + j] = hi ? t1p.u[j] : rcv;
    }

    // ---- O^T += V * P^T : preload all 8 A-fragments (distinct regs -> clause
    //      of ds_load_b128, partial dscnt waits instead of 8 serialized waits)
    V16 av0[4], av1[4];
    #pragma unroll
    for (int dt = 0; dt < 4; ++dt) {
      const __bf16* vr = myV + (size_t)(dt * 16) * VSTR;
      av0[dt].u[0] = *(const u32x4*)(vr + kbase);
      av0[dt].u[1] = *(const u32x4*)(vr + kbase + 8);
    }
    #pragma unroll
    for (int dt = 0; dt < 4; ++dt) {
      const __bf16* vr = myV + (size_t)((dt + 4) * 16) * VSTR;
      av1[dt].u[0] = *(const u32x4*)(vr + kbase);
      av1[dt].u[1] = *(const u32x4*)(vr + kbase + 8);
    }
    #pragma unroll
    for (int dt = 0; dt < 4; ++dt)
      o[dt] = __builtin_amdgcn_wmma_f32_16x16x32_bf16(false, av0[dt].v, false, bp.v,
                                                      (short)0, o[dt], false, false);
    #pragma unroll
    for (int dt = 0; dt < 4; ++dt)
      o[dt + 4] = __builtin_amdgcn_wmma_f32_16x16x32_bf16(false, av1[dt].v, false, bp.v,
                                                          (short)0, o[dt + 4], false, false);

    __syncthreads();   // single barrier: publishes chunk it+1, retires reads of it
  }

  // ---- merge the two Z-halves per M-group (overlay in buffer 0)
  float* oRed = (float*)(smem + ORED_OFF);
  float* mRed = (float*)(smem + MRED_OFF);
  float* lRed = (float*)(smem + LRED_OFF);

  if (zg == 1) {
    float* obase = oRed + mg * (MTILE * OSTR) + col * OSTR + 8 * hi;
    #pragma unroll
    for (int dt = 0; dt < 8; ++dt) {
      f32x4 lo  = {o[dt][0], o[dt][1], o[dt][2], o[dt][3]};
      f32x4 hi4 = {o[dt][4], o[dt][5], o[dt][6], o[dt][7]};
      *(f32x4*)(obase + dt * 16)     = lo;
      *(f32x4*)(obase + dt * 16 + 4) = hi4;
    }
    if (hi == 0) {
      mRed[mg * MTILE + col] = mrow;
      lRed[mg * MTILE + col] = lrow;
    }
  }
  __syncthreads();

  if (zg == 0) {
    const float mb = mRed[mg * MTILE + col];
    const float lb = lRed[mg * MTILE + col];
    const float mnew = fmaxf(mrow, mb);
    const float sa = __expf(mrow - mnew);
    const float sb = __expf(mb - mnew);
    const float linv = 1.0f / (lrow * sa + lb * sb);
    const float* obase = oRed + mg * (MTILE * OSTR) + col * OSTR + 8 * hi;
    float* orow = out + (size_t)(t0 + col) * MEMDIM + head * HD + 8 * hi;
    #pragma unroll
    for (int dt = 0; dt < 8; ++dt) {
      f32x4 b0 = *(const f32x4*)(obase + dt * 16);
      f32x4 b1 = *(const f32x4*)(obase + dt * 16 + 4);
      f32x4 r0, r1;
      r0.x = (o[dt][0]*sa + b0.x*sb)*linv; r0.y = (o[dt][1]*sa + b0.y*sb)*linv;
      r0.z = (o[dt][2]*sa + b0.z*sb)*linv; r0.w = (o[dt][3]*sa + b0.w*sb)*linv;
      r1.x = (o[dt][4]*sa + b1.x*sb)*linv; r1.y = (o[dt][5]*sa + b1.y*sb)*linv;
      r1.z = (o[dt][6]*sa + b1.z*sb)*linv; r1.w = (o[dt][7]*sa + b1.w*sb)*linv;
      *(f32x4*)(orow + dt * 16)     = r0;
      *(f32x4*)(orow + dt * 16 + 4) = r1;
    }
  }
}

extern "C" void kernel_launch(void* const* d_in, const int* in_sizes, int n_in,
                              void* d_out, int out_size, void* d_ws, size_t ws_size,
                              hipStream_t stream) {
  (void)in_sizes; (void)n_in; (void)out_size; (void)d_ws; (void)ws_size;
  const float* h        = (const float*)d_in[0];
  const float* keys_w   = (const float*)d_in[1];
  const float* values_w = (const float*)d_in[2];
  float* out = (float*)d_out;

  dim3 grid(16 * NHEADS);   // 16 token-tiles per head; head is the slow dim
  dim3 block(32 * NWAVES);  // 256 threads = 8 wave32
  kvmem_attn_kernel<<<grid, block, 0, stream>>>(h, keys_w, values_w, out);
}
